// GraphSAGE_79998060855855
// MI455X (gfx1250) — compile-verified, matrix operand order checked by reading, not run on hardware
//
#include <hip/hip_runtime.h>

// ---------------------------------------------------------------------------
// GraphSAGE (3-layer, mean-agg SAGEConv + BatchNorm + ReLU) for gfx1250.
// GEMMs use v_wmma_f32_16x16x32_bf16 (f32 accumulate) with register blocking
// over all output-column tiles (A-tiles loaded once per K-step, reused for
// every column tile). Aggregation uses coalesced float4 gathers +
// global_atomic_add_f32 scatters (one wave32 per edge).
// ---------------------------------------------------------------------------

typedef __bf16 bf16;
typedef __attribute__((ext_vector_type(8)))  __bf16 v8bf;
typedef __attribute__((ext_vector_type(16))) __bf16 v16bf;
typedef __attribute__((ext_vector_type(8)))  float  v8f;

#define BN_EPS 1e-5f

// ---------------- utility kernels ----------------

__global__ void k_fill_zero(float* p, long n) {
    long i = (long)blockIdx.x * blockDim.x + threadIdx.x;
    if (i < n) p[i] = 0.0f;
}

__global__ void k_cvt_f32_bf16(const float* in, bf16* out, long n) {
    long i = (long)blockIdx.x * blockDim.x + threadIdx.x;
    if (i < n) out[i] = (bf16)in[i];
}

// ---------------- degree ----------------

__global__ void k_degree(const int* ei, float* deg, int E) {
    int e = blockIdx.x * blockDim.x + threadIdx.x;
    if (e < E) atomicAdd(&deg[ei[E + e]], 1.0f);
}

__global__ void k_inv_degree(float* deg, int n) {
    int i = blockIdx.x * blockDim.x + threadIdx.x;
    if (i < n) deg[i] = 1.0f / fmaxf(deg[i], 1.0f);
}

// ---------------- edge scatter (mean numerator), C = 128 fixed ----------------
// One wave32 per edge; lane handles 4 contiguous features (float4 gather,
// 4x global_atomic_add_f32 scatter).

__global__ void k_scatter(const int* __restrict__ ei, const float* __restrict__ x,
                          float* __restrict__ agg, int E) {
    long gid  = (long)blockIdx.x * blockDim.x + threadIdx.x;
    int  e    = (int)(gid >> 5);
    int  lane = (int)(gid & 31);
    if (e >= E) return;
    int s = ei[e];
    int d = ei[E + e];
    float4 v = ((const float4*)(x + (size_t)s * 128))[lane];
    float* o = agg + (size_t)d * 128 + lane * 4;
    atomicAdd(o + 0, v.x);
    atomicAdd(o + 1, v.y);
    atomicAdd(o + 2, v.z);
    atomicAdd(o + 3, v.w);
}

// ---------------- agg /= deg, convert to bf16 (C = 128) ----------------

__global__ void k_scale_cvt(const float* __restrict__ agg, const float* __restrict__ invd,
                            bf16* __restrict__ out, int nrows) {
    long i = (long)blockIdx.x * blockDim.x + threadIdx.x;
    if (i >= (long)nrows * 128) return;
    int n = (int)(i >> 7);
    out[i] = (bf16)(agg[i] * invd[n]);
}

// ---------------- WMMA GEMM: Z = Aagg @ Wl^T + bias + Aroot @ Wr^T ----------
// A matrices: [nrows, Cin] bf16 row-major.  W: [Cout, Cin] bf16 row-major
// (so B = W^T is loaded by reading W rows contiguously).
// One wave owns a 16-row stripe and ALL NT = Cout/16 column tiles, register
// blocking NT accumulators so each A-tile pair is loaded once per K-step.
// ISA 16-bit A 16x32 layout: lane<16 -> row=lane, K in {k0..k0+7, k0+16..k0+23};
// lane>=16 -> row=lane-16, K in {k0+8..k0+15, k0+24..k0+31}.

__device__ __forceinline__ v16bf load_tile16x32(const bf16* p) {
    v8bf lo = *(const v8bf*)p;
    v8bf hi = *(const v8bf*)(p + 16);
    return __builtin_shufflevector(lo, hi, 0, 1, 2, 3, 4, 5, 6, 7,
                                   8, 9, 10, 11, 12, 13, 14, 15);
}

template <int NT>
__global__ void k_sage_gemm(const bf16* __restrict__ Aagg, const bf16* __restrict__ Aroot,
                            const bf16* __restrict__ Wl,   const bf16* __restrict__ Wr,
                            const float* __restrict__ bias, float* __restrict__ Z,
                            int Cin, int mtiles) {
    const int Cout = NT * 16;
    int wave = blockIdx.x * (blockDim.x >> 5) + (threadIdx.x >> 5);
    if (wave >= mtiles) return;                   // wave-uniform: EXEC stays all-1s

    int lane  = threadIdx.x & 31;
    int rsel  = lane & 15;
    int khalf = (lane >> 4) & 1;                  // 0: K {0..7,16..23}, 1: K {8..15,24..31}

    const bf16* a1base = Aagg  + (size_t)(wave * 16 + rsel) * Cin + khalf * 8;
    const bf16* a2base = Aroot + (size_t)(wave * 16 + rsel) * Cin + khalf * 8;
    const bf16* b1base = Wl    + (size_t)rsel * Cin + khalf * 8;
    const bf16* b2base = Wr    + (size_t)rsel * Cin + khalf * 8;

    v8f acc[NT] = {};
#pragma unroll
    for (int k0 = 0; k0 < 128; k0 += 32) {
        if (k0 >= Cin) break;
        v16bf A1 = load_tile16x32(a1base + k0);   // loaded once, reused for all NT tiles
        v16bf A2 = load_tile16x32(a2base + k0);
#pragma unroll
        for (int nt = 0; nt < NT; ++nt) {
            size_t woff = (size_t)nt * 16 * Cin + k0;
            v16bf B1 = load_tile16x32(b1base + woff);
            v16bf B2 = load_tile16x32(b2base + woff);
            acc[nt] = __builtin_amdgcn_wmma_f32_16x16x32_bf16(
                false, A1, false, B1, (short)0, acc[nt], false, false);
            acc[nt] = __builtin_amdgcn_wmma_f32_16x16x32_bf16(
                false, A2, false, B2, (short)0, acc[nt], false, false);
        }
    }

    // C/D layout: lane<16 -> col=lane, rows v+0..7; lane>=16 -> col=lane-16, rows v+8..15
    int colbase = lane & 15;
    int rowbase = wave * 16 + ((lane >> 4) << 3);
#pragma unroll
    for (int nt = 0; nt < NT; ++nt) {
        int col = nt * 16 + colbase;
        float b = bias[col];
#pragma unroll
        for (int v = 0; v < 8; ++v)
            Z[(size_t)(rowbase + v) * Cout + col] = acc[nt][v] + b;
    }
}

// ---------------- BatchNorm statistics ----------------

__global__ void k_bn_stats(const float* __restrict__ z, float* __restrict__ stats,
                           int nrows, int C) {
    int t      = threadIdx.x;
    int cpt    = blockDim.x / C;                  // rows handled per block per step
    int c      = t % C;
    int r      = blockIdx.x * cpt + t / C;
    int stride = gridDim.x * cpt;
    float s = 0.0f, q = 0.0f;
    for (; r < nrows; r += stride) {
        float v = z[(size_t)r * C + c];
        s += v;
        q += v * v;
    }
    atomicAdd(&stats[c], s);
    atomicAdd(&stats[C + c], q);
}

__global__ void k_bn_finalize(const float* __restrict__ stats,
                              const float* __restrict__ gamma, const float* __restrict__ beta,
                              float* __restrict__ ss, int C, float invN) {
    int c = threadIdx.x;
    if (c >= C) return;
    float mean = stats[c] * invN;
    float var  = stats[C + c] * invN - mean * mean;
    float sc   = gamma[c] * rsqrtf(var + BN_EPS);
    ss[c]      = sc;
    ss[C + c]  = beta[c] - mean * sc;
}

__global__ void k_bn_apply(float* __restrict__ z, const float* __restrict__ ss,
                           bf16* __restrict__ hbf, long total, int C, int relu) {
    long i = (long)blockIdx.x * blockDim.x + threadIdx.x;
    if (i >= total) return;
    int c = (int)(i % C);
    float v = z[i] * ss[c] + ss[C + c];
    if (relu) v = fmaxf(v, 0.0f);
    z[i] = v;
    if (hbf) hbf[i] = (bf16)v;
}

// ---------------- host orchestration ----------------

extern "C" void kernel_launch(void* const* d_in, const int* in_sizes, int n_in,
                              void* d_out, int out_size, void* d_ws, size_t ws_size,
                              hipStream_t stream) {
    (void)n_in; (void)out_size; (void)ws_size;

    const float* x  = (const float*)d_in[0];
    const int*   ei = (const int*)d_in[1];
    const float* Wl[3]    = {(const float*)d_in[2],  (const float*)d_in[7],  (const float*)d_in[12]};
    const float* bl[3]    = {(const float*)d_in[3],  (const float*)d_in[8],  (const float*)d_in[13]};
    const float* Wr[3]    = {(const float*)d_in[4],  (const float*)d_in[9],  (const float*)d_in[14]};
    const float* gamma[3] = {(const float*)d_in[5],  (const float*)d_in[10], (const float*)d_in[15]};
    const float* beta[3]  = {(const float*)d_in[6],  (const float*)d_in[11], (const float*)d_in[16]};

    const int N = in_sizes[0] / 128;
    const int E = in_sizes[1] / 2;
    const int cin[3]  = {128, 128, 128};
    const int cout[3] = {128, 128, 64};

    // workspace carve-up (256B aligned slices)
    char*  ws  = (char*)d_ws;
    size_t off = 0;
    auto take = [&](size_t bytes) -> void* {
        void* p = ws + off;
        off += (bytes + 255) & ~(size_t)255;
        return p;
    };
    float* inv_deg = (float*)take((size_t)N * 4);
    float* agg     = (float*)take((size_t)N * 128 * 4);
    float* hA      = (float*)take((size_t)N * 128 * 4);
    float* hB      = (float*)take((size_t)N * 128 * 4);
    bf16*  agg_bf  = (bf16*) take((size_t)N * 128 * 2);
    bf16*  h_bf    = (bf16*) take((size_t)N * 128 * 2);
    bf16*  wl_bf[3]; bf16* wr_bf[3];
    for (int l = 0; l < 3; ++l) {
        wl_bf[l] = (bf16*)take((size_t)cout[l] * cin[l] * 2);
        wr_bf[l] = (bf16*)take((size_t)cout[l] * cin[l] * 2);
    }
    float* stats = (float*)take(2 * 128 * 4);
    float* ss    = (float*)take(2 * 128 * 4);

    const int TB = 256;
    auto blocks = [](long n, int tb) { return (unsigned)((n + tb - 1) / tb); };

    // 1) weights + x -> bf16
    for (int l = 0; l < 3; ++l) {
        long wn = (long)cout[l] * cin[l];
        k_cvt_f32_bf16<<<blocks(wn, TB), TB, 0, stream>>>(Wl[l], wl_bf[l], wn);
        k_cvt_f32_bf16<<<blocks(wn, TB), TB, 0, stream>>>(Wr[l], wr_bf[l], wn);
    }
    k_cvt_f32_bf16<<<blocks((long)N * 128, TB), TB, 0, stream>>>(x, h_bf, (long)N * 128);

    // 2) degree -> inverse degree (same for all layers)
    k_fill_zero<<<blocks(N, TB), TB, 0, stream>>>(inv_deg, N);
    k_degree<<<blocks(E, TB), TB, 0, stream>>>(ei, inv_deg, E);
    k_inv_degree<<<blocks(N, TB), TB, 0, stream>>>(inv_deg, N);

    // 3) layers
    const float* in_f    = x;
    float*       outs[3] = {hA, hB, (float*)d_out};
    for (int l = 0; l < 3; ++l) {
        float* out = outs[l];
        const int C = cout[l];

        k_fill_zero<<<blocks((long)N * 128, TB), TB, 0, stream>>>(agg, (long)N * 128);
        k_scatter<<<blocks((long)E * 32, TB), TB, 0, stream>>>(ei, in_f, agg, E);
        k_scale_cvt<<<blocks((long)N * 128, TB), TB, 0, stream>>>(agg, inv_deg, agg_bf, N);

        int mtiles = N / 16;                // N = 100000 is an exact multiple of 16
        unsigned gblocks = blocks(mtiles, 8);   // 8 waves (one stripe each) per 256-thread block
        if (C == 128) {
            k_sage_gemm<8><<<gblocks, TB, 0, stream>>>(
                agg_bf, h_bf, wl_bf[l], wr_bf[l], bl[l], out, cin[l], mtiles);
        } else {
            k_sage_gemm<4><<<gblocks, TB, 0, stream>>>(
                agg_bf, h_bf, wl_bf[l], wr_bf[l], bl[l], out, cin[l], mtiles);
        }

        k_fill_zero<<<1, 256, 0, stream>>>(stats, 2 * C);
        k_bn_stats<<<512, TB, 0, stream>>>(out, stats, N, C);
        k_bn_finalize<<<1, 128, 0, stream>>>(stats, gamma[l], beta[l], ss, C, 1.0f / (float)N);
        k_bn_apply<<<blocks((long)N * C, TB), TB, 0, stream>>>(
            out, ss, (l < 2) ? h_bf : (bf16*)nullptr, (long)N * C, C, (l < 2) ? 1 : 0);

        in_f = out;
    }
}
